// FP8PalettizedConv2d_59889023975691
// MI455X (gfx1250) — matrix operand order repaired and checked by reading, not compiled, not run on hardware
//
#include <hip/hip_runtime.h>

typedef __attribute__((ext_vector_type(16))) __bf16 v16bf;
typedef __attribute__((ext_vector_type(8)))  __bf16 v8bf;
typedef __attribute__((ext_vector_type(8)))  float  v8f;

#define CIN    128
#define HDIM   112
#define WDIM   112
#define COUT   256
#define BATCH  16
#define KSTEPS 36            // 1152 / 32
#define NTILE_W 7            // 112 / 16
#define HPAIRS  56           // 112 / 2
#define NBLOCKS (BATCH * HPAIRS * NTILE_W)   // 6272
#define LDS_COLS 18          // 16 + 2 halo
#define LDS_ROWS 4           // h0-1 .. h0+2
#define CH_PAD   136         // 128 channels + 8 pad halfs (272B/col -> 4-bank shift)
#define CELLS    (LDS_ROWS * LDS_COLS)       // 72

// ------------------------------------------------------------------
// Prep: dequantize palettized weights straight into WMMA A-fragment
// layout (bf16):  ws[mtile(16)][kstep(36)][lane(32)][e(16)]
// A 16x32 bf16 layout (contiguous K, matches B): lanes 0-15 hold M=lane,
//   e0..15 -> K=e ; lanes 16-31 hold M=lane-16,
//   e0..15 -> K=16+e.
// K ordering: k = (kh*3+kw)*128 + cin
// ------------------------------------------------------------------
__global__ void dequant_wfrag_kernel(const int* __restrict__ widx,
                                     const float* __restrict__ lut,
                                     __bf16* __restrict__ wfrag) {
  int t = blockIdx.x * blockDim.x + threadIdx.x;
  if (t >= 16 * KSTEPS * 32 * 16) return;
  int e     = t & 15;
  int lane  = (t >> 4) & 31;
  int kstep = (t >> 9) % KSTEPS;
  int mtile = t / (512 * KSTEPS);

  int koff = e + (lane >= 16 ? 16 : 0);
  int k    = kstep * 32 + koff;
  int p    = k >> 7;          // kh*3 + kw
  int cin  = k & 127;
  int kh   = p / 3;
  int kw   = p % 3;
  int oc   = mtile * 16 + (lane & 15);

  int idx = widx[((oc * CIN + cin) * 3 + kh) * 3 + kw];
  wfrag[t] = (__bf16)lut[idx];
}

// ------------------------------------------------------------------
// Implicit-GEMM conv, LDS-staged inputs (channel-contiguous layout so
// B fragments are 2 x ds_load_b128), 2M x 2N register blocking.
// Block: one (b, h-pair, 16-wide w tile); 8 waves x 2 mtiles = 256 oc.
// ------------------------------------------------------------------
__global__ void __launch_bounds__(256)
palconv_wmma_kernel(const float* __restrict__ in,
                    const __bf16* __restrict__ wfrag,
                    const float* __restrict__ bias,
                    float* __restrict__ out) {
  // layout: smem[(row*18 + col)*136 + ch]  (ch 0..127 used, 128..135 pad)
  __shared__ __bf16 smem[CELLS * CH_PAD];    // 19,584 B

  const int bx = blockIdx.x;
  const int wt = bx % NTILE_W;
  const int hp = (bx / NTILE_W) % HPAIRS;
  const int b  = bx / (NTILE_W * HPAIRS);
  const int h0 = hp * 2;
  const int w0 = wt * 16;

  // ---- Stage input patch -> LDS bf16 (transposed to channel-contiguous),
  //      zero-filled halo: hot loop needs no bounds checks.
  for (int s = threadIdx.x; s < CELLS * CIN; s += 256) {
    int col = s % LDS_COLS;                 // global-coalesced walk
    int t2  = s / LDS_COLS;
    int row = t2 & 3;
    int ch  = t2 >> 2;
    int ihh = h0 - 1 + row;
    int iww = w0 - 1 + col;
    float v = 0.0f;
    if (ihh >= 0 && ihh < HDIM && iww >= 0 && iww < WDIM)
      v = in[(((size_t)b * CIN + ch) * HDIM + ihh) * WDIM + iww];
    smem[(row * LDS_COLS + col) * CH_PAD + ch] = (__bf16)v;
  }
  __syncthreads();

  const int lane = threadIdx.x & 31;
  const int wave = threadIdx.x >> 5;
  const int m0 = wave;        // mtiles: wave and wave+8
  const int m1 = wave + 8;

  const int  n   = lane & 15;
  const bool hi  = lane >= 16;
  const int  hig = hi ? 16 : 0;

  v8f a00 = {}, a01 = {}, a10 = {}, a11 = {};

  for (int p = 0; p < 9; ++p) {
    const int kh = p / 3;
    const int kw = p % 3;
    const __bf16* cell0 = smem + (kh * LDS_COLS + n + kw) * CH_PAD + hig;

    // prefetch next K-block of the streamed A fragments (L2 hint)
    if (p < 8) {
      __builtin_prefetch(wfrag + (((size_t)m0 * KSTEPS + (p + 1) * 4) * 32) * 16, 0, 1);
      __builtin_prefetch(wfrag + (((size_t)m1 * KSTEPS + (p + 1) * 4) * 32) * 16, 0, 1);
    }

#pragma unroll
    for (int cb = 0; cb < 4; ++cb) {
      const int kstep = p * 4 + cb;
      const v16bf am0 = *(const v16bf*)(wfrag +
          (((size_t)m0 * KSTEPS + kstep) * 32 + lane) * 16);
      const v16bf am1 = *(const v16bf*)(wfrag +
          (((size_t)m1 * KSTEPS + kstep) * 32 + lane) * 16);

      // B fragments: 16 consecutive channels -> 2 x ds_load_b128 each
      const __bf16* bs0 = cell0 + cb * 32;                      // output row h0
      const __bf16* bs1 = bs0 + LDS_COLS * CH_PAD;              // output row h1
      v8bf b0l = *(const v8bf*)(bs0);
      v8bf b0h = *(const v8bf*)(bs0 + 8);
      v8bf b1l = *(const v8bf*)(bs1);
      v8bf b1h = *(const v8bf*)(bs1 + 8);
      v16bf b0 = __builtin_shufflevector(b0l, b0h,
          0, 1, 2, 3, 4, 5, 6, 7, 8, 9, 10, 11, 12, 13, 14, 15);
      v16bf b1 = __builtin_shufflevector(b1l, b1h,
          0, 1, 2, 3, 4, 5, 6, 7, 8, 9, 10, 11, 12, 13, 14, 15);

      a00 = __builtin_amdgcn_wmma_f32_16x16x32_bf16(false, am0, false, b0,
                                                    (short)0, a00, false, false);
      a01 = __builtin_amdgcn_wmma_f32_16x16x32_bf16(false, am0, false, b1,
                                                    (short)0, a01, false, false);
      a10 = __builtin_amdgcn_wmma_f32_16x16x32_bf16(false, am1, false, b0,
                                                    (short)0, a10, false, false);
      a11 = __builtin_amdgcn_wmma_f32_16x16x32_bf16(false, am1, false, b1,
                                                    (short)0, a11, false, false);
    }
  }

  // C/D layout: lane<16 -> M=r, lane>=16 -> M=8+r ; N = lane&15
  const int oc0 = m0 * 16 + (hi ? 8 : 0);
  const int oc1 = m1 * 16 + (hi ? 8 : 0);
  const size_t w = w0 + n;
  const size_t HW = (size_t)HDIM * WDIM;
  float* d00 = out + (((size_t)b * COUT + oc0) * HDIM + h0) * WDIM + w;
  float* d10 = out + (((size_t)b * COUT + oc1) * HDIM + h0) * WDIM + w;
#pragma unroll
  for (int r = 0; r < 8; ++r) {
    const float bv0 = bias[oc0 + r];
    const float bv1 = bias[oc1 + r];
    d00[r * HW]        = a00[r] + bv0;   // (m0, h0)
    d00[r * HW + WDIM] = a01[r] + bv0;   // (m0, h1)
    d10[r * HW]        = a10[r] + bv1;   // (m1, h0)
    d10[r * HW + WDIM] = a11[r] + bv1;   // (m1, h1)
  }
}

extern "C" void kernel_launch(void* const* d_in, const int* in_sizes, int n_in,
                              void* d_out, int out_size, void* d_ws, size_t ws_size,
                              hipStream_t stream) {
  const float* input = (const float*)d_in[0];
  const int*   widx  = (const int*)d_in[1];
  const float* lut   = (const float*)d_in[2];
  const float* bias  = (const float*)d_in[3];
  float*  outp  = (float*)d_out;
  __bf16* wfrag = (__bf16*)d_ws;   // needs 16*36*32*16*2 = 589,824 bytes

  const int total = 16 * KSTEPS * 32 * 16;
  dequant_wfrag_kernel<<<(total + 255) / 256, 256, 0, stream>>>(widx, lut, wfrag);

  palconv_wmma_kernel<<<NBLOCKS, 256, 0, stream>>>(input, wfrag, bias, outp);
}
